// Pointwise_80719615361307
// MI455X (gfx1250) — matrix operand order repaired and checked by reading, not compiled, not run on hardware
//
#include <hip/hip_runtime.h>

typedef __attribute__((ext_vector_type(16))) _Float16     v16h;
typedef __attribute__((ext_vector_type(8)))  _Float16     v8h;
typedef __attribute__((ext_vector_type(8)))  float        v8f;
typedef __attribute__((ext_vector_type(4)))  float        v4f;
typedef __attribute__((ext_vector_type(8)))  unsigned int v8ui;

namespace {
constexpr int kB     = 64;
constexpr int kCin   = 64;
constexpr int kCout  = 64;
constexpr int kH     = 112, kW = 112;
constexpr int kHW    = kH * kW;     // 12544
constexpr int kSTile = 64;          // spatial positions staged per iteration
constexpr int kIter  = 4;           // staged tiles per workgroup (49 blocks/image)
constexpr int kCP    = 72;          // LDS row stride in halves (conflict-free b128 reads)
}

// sign() on 16 packed halves as pure bit ops: one bfi/and_or per half-pair.
// f32->f16 RTNE preserves the sign bit, so (sign|1.0h) reproduces the true
// sign(x)=+-1 even for tiny x that flushed to +-0.0h. Only an exactly-zero
// f32 input (measure-zero under jax.random.normal) deviates (gives +-1, not 0).
__device__ __forceinline__ v16h sign16(v16h v) {
  v8ui u = __builtin_bit_cast(v8ui, v);
  v8ui r;
#pragma unroll
  for (int i = 0; i < 8; ++i)
    r[i] = (u[i] & 0x80008000u) | 0x3C003C00u;   // v_and_or_b32 / v_bfi_b32
  return __builtin_bit_cast(v16h, r);
}

// out[b,o,s] = sum_c sign(x[b,c,s])*w[o,c] + x[b,c,s]*sign(w[o,c])
// GEMM view: D[o][s] (16x16 tiles) = A[o][k] * B[k][s], K = 128:
//   k <  64 : A = w (raw),   B = sign(x)
//   k >= 64 : A = sign(w),   B = x (raw)
__global__ __launch_bounds__(256)
void bconv1x1_wmma(const float* __restrict__ x,
                   const float* __restrict__ w,
                   float* __restrict__ out) {
  __shared__ _Float16 lx[kSTile * kCP];  // x tile, f16, layout [s][c]

  const int  t    = threadIdx.x;
  const int  lane = t & 31;
  const int  wave = t >> 5;        // 8 waves (wave32)
  const int  lrow = lane & 15;
  const bool hi   = lane >= 16;

  const int batch = blockIdx.y;
  const int sBlk  = blockIdx.x * (kSTile * kIter);

  const float* xb   = x   + (size_t)batch * kCin  * kHW;
  float*       outb = out + (size_t)batch * kCout * kHW;

  // ---- Weight A-fragments, built once per workgroup ----
  // 16KB table, re-read by every workgroup -> keep regular-temporal (L2-resident).
  // 16-bit A 16x32 layout: lane row M = lane%16;
  // half h -> k = (h/8)*16 + (lane>=16 ? 8 : 0) + h%8.
  const int o0 = (wave & 3) * 16;
  v16h wraw[2], wsgn[2];
  {
    const float* wrow = w + (size_t)(o0 + lrow) * kCin;
#pragma unroll
    for (int f = 0; f < 2; ++f) {          // input-channel halves [0,32),[32,64)
      v16h a;
#pragma unroll
      for (int g = 0; g < 2; ++g) {
        const int c0 = f * 32 + g * 16 + (hi ? 8 : 0);
        const v4f p  = *(const v4f*)(wrow + c0);
        const v4f q  = *(const v4f*)(wrow + c0 + 4);
        a[g * 8 + 0] = (_Float16)p.x; a[g * 8 + 1] = (_Float16)p.y;
        a[g * 8 + 2] = (_Float16)p.z; a[g * 8 + 3] = (_Float16)p.w;
        a[g * 8 + 4] = (_Float16)q.x; a[g * 8 + 5] = (_Float16)q.y;
        a[g * 8 + 6] = (_Float16)q.z; a[g * 8 + 7] = (_Float16)q.w;
      }
      wraw[f] = a;
      wsgn[f] = sign16(a);
    }
  }

  const int swv = (wave >> 2) * 16;  // wave's first spatial sub-tile (0 / 16)

  for (int it = 0; it < kIter; ++it) {
    const int s0 = sBlk + it * kSTile;

    if (it) __syncthreads();  // LDS reuse across iterations

    // ---- Stage x tile: 64 ch x 64 spatial, f32 -> f16, to LDS[s][c] ----
    // x is streamed exactly once per call -> non-temporal loads (don't
    // pollute WGP$/L2; weights stay cached instead).
    // 4-lane groups read 64B-contiguous runs of one channel row.
    {
      const int    c     = t >> 2;
      const int    inner = t & 3;
      const float* src   = xb + (size_t)c * kHW + s0 + inner * 4;
#pragma unroll
      for (int j = 0; j < 4; ++j) {
        const v4f v = __builtin_nontemporal_load((const v4f*)(src + j * 16));
        const int s = inner * 4 + j * 16;
        lx[(s + 0) * kCP + c] = (_Float16)v.x;
        lx[(s + 1) * kCP + c] = (_Float16)v.y;
        lx[(s + 2) * kCP + c] = (_Float16)v.z;
        lx[(s + 3) * kCP + c] = (_Float16)v.w;
      }
    }
    __syncthreads();

#pragma unroll
    for (int st = 0; st < 2; ++st) {
      const int sSub = swv + st * 32;  // waves cover spatial offsets 0,16,32,48

      // ---- B-fragments from LDS ----
      // 16-bit B 32x16 layout: lane col N = lane%16;
      // half h -> k = (h/8)*16 + (lane>=16 ? 8 : 0) + h%8.
      v16h braw[2], bsgn[2];
#pragma unroll
      for (int f = 0; f < 2; ++f) {
        const int       cStart = f * 32 + (hi ? 8 : 0);
        const _Float16* p      = &lx[(sSub + lrow) * kCP + cStart];
        union { v16h v; v8h h[2]; } u;
        u.h[0]  = *(const v8h*)(p);      // ds_load_b128
        u.h[1]  = *(const v8h*)(p + 16); // ds_load_b128
        braw[f] = u.v;
        bsgn[f] = sign16(u.v);
      }

      // ---- K=128 accumulation: 4 x v_wmma_f32_16x16x32_f16 ----
      v8f acc = {};
      acc = __builtin_amdgcn_wmma_f32_16x16x32_f16(false, wraw[0], false, bsgn[0],
                                                   (short)0, acc, false, false);
      acc = __builtin_amdgcn_wmma_f32_16x16x32_f16(false, wraw[1], false, bsgn[1],
                                                   (short)0, acc, false, false);
      acc = __builtin_amdgcn_wmma_f32_16x16x32_f16(false, wsgn[0], false, braw[0],
                                                   (short)0, acc, false, false);
      acc = __builtin_amdgcn_wmma_f32_16x16x32_f16(false, wsgn[1], false, braw[1],
                                                   (short)0, acc, false, false);

      // ---- Store: C/D layout lane(0-15)=16 contiguous spatial floats ----
      // acc[r] @ lane L: o = o0 + r + (hi?8:0), s = s0 + sSub + (L%16)
      // Output is write-once/stream -> non-temporal stores.
      const int sg   = s0 + sSub + lrow;
      float*    orow = outb + (size_t)(o0 + (hi ? 8 : 0)) * kHW + sg;
#pragma unroll
      for (int r = 0; r < 8; ++r)
        __builtin_nontemporal_store(acc[r], &orow[(size_t)r * kHW]);
    }
  }
}

extern "C" void kernel_launch(void* const* d_in, const int* in_sizes, int n_in,
                              void* d_out, int out_size, void* d_ws, size_t ws_size,
                              hipStream_t stream) {
  (void)in_sizes; (void)n_in; (void)out_size; (void)d_ws; (void)ws_size;
  const float* x = (const float*)d_in[0];       // [64,64,112,112] f32
  const float* w = (const float*)d_in[1];       // [64,64,1,1] f32
  // d_in[2], d_in[3]: alpha_x, alpha_w — only affect backward; unused in fwd.
  float* out = (float*)d_out;                   // [64,64,112,112] f32

  dim3 grid(kHW / (kSTile * kIter), kB);        // (49, 64)
  bconv1x1_wmma<<<grid, 256, 0, stream>>>(x, w, out);
}
